// ArFSSM_77103252898304
// MI455X (gfx1250) — matrix operand order repaired and checked by reading, not compiled, not run on hardware
//
#include <hip/hip_runtime.h>
#include <hip/hip_bf16.h>

typedef __attribute__((ext_vector_type(16))) _Float16 v16h;
typedef __attribute__((ext_vector_type(8)))  float    v8f;

// Problem constants (setup_inputs): B=256, T=1000, DX=DZ=100, H1=H2=10, DDX=DDZ=20
#define B_TOT 256
#define T_LEN 1000
#define DXC   100
#define DZC   100
#define H1C   10
#define DDXC  20
#define DDZC  20

// B-fragment table (each fragment = 512 halves = 1KB in LDS)
#define F_WT1  0   // 2 frags  (N=20,  K=10)
#define F_WLOC 2   // 7 frags  (N=100, K=20)
#define F_WSC  9   // 7 frags  (N=100, K=20)
#define F_WIH1 16  // 8 frags  (N=30,  K=100) [s*2+tn]
#define F_WHH1 24  // 2 frags  (N=30,  K=10)
#define F_WIH2 26  // 8 frags  (N=30,  K=100)
#define F_WHH2 34  // 2 frags  (N=30,  K=10)
#define F_WE1  36  // 2 frags  (N=20,  K=20)
#define F_WE2  38  // 2 frags  (N=20,  K=20)
#define F_WE3  40  // 7 frags  (N=100, K=20)
#define NFRAG  47

__device__ __forceinline__ v8f splat8(float x) { v8f v = {x,x,x,x,x,x,x,x}; return v; }

__device__ __forceinline__ v8f wmma16(v16h a, v16h b, v8f c) {
  // (neg_a, A, neg_b, B, c_mod, C, reuse_a, reuse_b)
  return __builtin_amdgcn_wmma_f32_16x16x32_f16(false, a, false, b, (short)0, c, false, false);
}

__device__ __forceinline__ float sigm(float x)  { return 1.0f / (1.0f + __expf(-x)); }
__device__ __forceinline__ float softp(float x) { return fmaxf(x, 0.0f) + __logf(1.0f + __expf(-fabsf(x))); }
__device__ __forceinline__ float tanh_s(float x) { float e = __expf(2.0f * x); return 1.0f - 2.0f / (e + 1.0f); }

// A operand (16x32 f16, ISA 7.12.2): lane holds row m=lane&15; halves 0..7 = K klo..klo+7,
// halves 8..15 = K klo+16..klo+23, klo = (lane<16)?0:8 (+32*kslice). Two ds_load_b128.
__device__ __forceinline__ v16h loadA(const _Float16* base, int strideH, int lane, int s) {
  const _Float16* row = base + (lane & 15) * strideH + s * 32 + ((lane < 16) ? 0 : 8);
  union { v16h h; uint4 u[2]; } r;
  r.u[0] = *(const uint4*)(row);
  r.u[1] = *(const uint4*)(row + 16);
  return r.h;
}

// B fragments pre-packed per-lane-contiguous: frag f, lane l -> 32 bytes at f*1024 + l*32
__device__ __forceinline__ v16h loadB(const _Float16* sB, int f, int lane) {
  union { v16h h; uint4 u[2]; } r;
  const uint4* p = (const uint4*)(sB + f * 512 + lane * 16);
  r.u[0] = p[0]; r.u[1] = p[1];
  return r.h;
}

// Pack W (N rows x K cols, row-major; op is act @ W.T so B[k][n] = W[n][k]) into
// B-fragment layout: lane n = (l&15)+16*tn holds K = 32*s + ((l<16)?0:8) + (j<8 ? j : j+8), j=0..15.
__device__ __forceinline__ void packB(const float* W, int N, int K, int s, int tn,
                                      _Float16* sB, int f, int lane) {
  int n  = (lane & 15) + 16 * tn;
  int kb = 32 * s + ((lane < 16) ? 0 : 8);
  _Float16* o = sB + f * 512 + lane * 16;
#pragma unroll
  for (int j = 0; j < 16; ++j) {
    int k = kb + ((j < 8) ? j : j + 8);
    float v = (n < N && k < K) ? W[n * K + k] : 0.0f;
    o[j] = (_Float16)v;
  }
}

// D layout (16x16 f32): lane holds col n=lane&15, rows mb..mb+7 with mb=(lane>>4)*8
__device__ __forceinline__ void storeD_h16(v8f a, _Float16* base, int strideH, int lane, bool relu) {
  int n = lane & 15, mb = (lane >> 4) * 8;
#pragma unroll
  for (int r = 0; r < 8; ++r) {
    float v = a[r];
    if (relu) v = fmaxf(v, 0.0f);
    base[(mb + r) * strideH + n] = (_Float16)v;
  }
}
__device__ __forceinline__ void storeD_f32(v8f a, float* base, int strideH, int lane) {
  int n = lane & 15, mb = (lane >> 4) * 8;
#pragma unroll
  for (int r = 0; r < 8; ++r) base[(mb + r) * strideH + n] = a[r];
}

// GRU combine: gates (r,z,n) at cols [0:10),[10:20),[20:30) in gI/gH (stride 33).
// 32 lanes cover 16 rows x 10 cols (5 cols per lane). Writes new h into f32 state,
// f16 A-staging, and the emitter concat buffer.
__device__ __forceinline__ void gruCombine(const float* gI, const float* gH, float* hf,
                                           _Float16* hs, _Float16* dxpart, int lane) {
  int m  = lane & 15;
  int cb = (lane >> 4) * 5;
#pragma unroll
  for (int i = 0; i < 5; ++i) {
    int c = cb + i;
    float rg = sigm(gI[m * 33 + c]      + gH[m * 33 + c]);
    float ug = sigm(gI[m * 33 + 10 + c] + gH[m * 33 + 10 + c]);
    float ng = tanh_s(gI[m * 33 + 20 + c] + rg * gH[m * 33 + 20 + c]);
    float hold = hf[m * 12 + c];
    float hnew = (1.0f - ug) * ng + ug * hold;
    hf[m * 12 + c] = hnew;
    _Float16 hh = (_Float16)hnew;
    hs[m * 40 + c]     = hh;
    dxpart[m * 40 + c] = hh;
  }
}

__global__ __launch_bounds__(32)
void ArFSSM_77103252898304_kernel(
    const float* __restrict__ x,     const float* __restrict__ eps,
    const float* __restrict__ W_ih1, const float* __restrict__ W_hh1,
    const float* __restrict__ b_ih1, const float* __restrict__ b_hh1,
    const float* __restrict__ W_ih2, const float* __restrict__ W_hh2,
    const float* __restrict__ b_ih2, const float* __restrict__ b_hh2,
    const float* __restrict__ h1_0,  const float* __restrict__ h2_0,
    const float* __restrict__ Wt1,   const float* __restrict__ bt1,
    const float* __restrict__ Wloc,  const float* __restrict__ bloc,
    const float* __restrict__ Wsc,   const float* __restrict__ bsc,
    const float* __restrict__ We1,   const float* __restrict__ be1,
    const float* __restrict__ We2,   const float* __restrict__ be2,
    const float* __restrict__ We3,   const float* __restrict__ be3,
    float* __restrict__ out)
{
  // Weights in B-fragment layout (47 KB) + activation staging. ~68 KB of the 320 KB WGP LDS.
  __shared__ __align__(16) _Float16 sB[NFRAG * 512];
  __shared__ __align__(16) _Float16 xs[16 * 136];   // x[t] tile, K padded to 128 (stride 272B)
  __shared__ __align__(16) _Float16 zs[16 * 136];   // z sample tile, K padded to 128
  __shared__ __align__(16) _Float16 hidS[16 * 40];  // relu(h1@Wt1.T) (K pad 32, stride 80B)
  __shared__ __align__(16) _Float16 h1s[16 * 40];   // h1 A-staging
  __shared__ __align__(16) _Float16 h2s[16 * 40];   // h2 A-staging
  __shared__ __align__(16) _Float16 dxs[16 * 40];   // concat(h1n, h2_old) A-staging
  __shared__ __align__(16) _Float16 es[16 * 40];    // emitter hidden A-staging
  __shared__ __align__(16) float    gI[16 * 33];    // gi gate buffer (stride 33 vs bank conflicts)
  __shared__ __align__(16) float    gH[16 * 33];    // gh gate buffer
  __shared__ __align__(16) float    h1f[16 * 12];   // f32 recurrent state copies
  __shared__ __align__(16) float    h2f[16 * 12];

  const int lane = threadIdx.x;
  const int n16  = lane & 15;
  const int mb   = (lane >> 4) * 8;
  const int m0   = blockIdx.x * 16;   // batch tile base

  // ---- init: zero staging (zero-padded K regions feed WMMA) ----
  for (int i = lane; i < 16 * 136; i += 32) { xs[i] = (_Float16)0.0f; zs[i] = (_Float16)0.0f; }
  for (int i = lane; i < 16 * 40;  i += 32) {
    h1s[i] = (_Float16)0.0f; h2s[i] = (_Float16)0.0f; dxs[i] = (_Float16)0.0f;
    es[i]  = (_Float16)0.0f; hidS[i] = (_Float16)0.0f;
  }

  // ---- pack all weight B-fragments into LDS (one time) ----
#pragma unroll
  for (int tn = 0; tn < 2; ++tn) {
    packB(Wt1,   DDZC, H1C, 0, tn, sB, F_WT1  + tn, lane);
    packB(W_hh1, 30,   H1C, 0, tn, sB, F_WHH1 + tn, lane);
    packB(W_hh2, 30,   H1C, 0, tn, sB, F_WHH2 + tn, lane);
    packB(We1,   DDXC, DDXC, 0, tn, sB, F_WE1 + tn, lane);
    packB(We2,   DDXC, DDXC, 0, tn, sB, F_WE2 + tn, lane);
  }
#pragma unroll
  for (int tn = 0; tn < 7; ++tn) {
    packB(Wloc, DZC, DDZC, 0, tn, sB, F_WLOC + tn, lane);
    packB(Wsc,  DZC, DDZC, 0, tn, sB, F_WSC  + tn, lane);
    packB(We3,  DXC, DDXC, 0, tn, sB, F_WE3  + tn, lane);
  }
#pragma unroll
  for (int s = 0; s < 4; ++s)
#pragma unroll
    for (int tn = 0; tn < 2; ++tn) {
      packB(W_ih1, 30, DZC, s, tn, sB, F_WIH1 + 2 * s + tn, lane);
      packB(W_ih2, 30, DXC, s, tn, sB, F_WIH2 + 2 * s + tn, lane);
    }

  // ---- recurrent state init ----
  if (lane < 16) {
#pragma unroll
    for (int c = 0; c < H1C; ++c) {
      float a = h1_0[c], b2 = h2_0[c];
      h1f[lane * 12 + c] = a;
      h2f[lane * 12 + c] = b2;
      h1s[lane * 40 + c] = (_Float16)a;
      h2s[lane * 40 + c] = (_Float16)b2;
      dxs[lane * 40 + 10 + c] = (_Float16)b2;   // emitter sees h2_shift (h2_0 at t=0)
    }
  }

  // ---- per-lane bias splats (C init values, indexed by column = n16 + 16*tile) ----
  float bLoc[7], bSc[7], bE3[7];
#pragma unroll
  for (int tn = 0; tn < 7; ++tn) {
    int col = n16 + 16 * tn;
    bLoc[tn] = (col < DZC) ? bloc[col] : 0.0f;
    bSc[tn]  = (col < DZC) ? bsc[col]  : 0.0f;
    bE3[tn]  = (col < DXC) ? be3[col]  : 0.0f;
  }
  float bT1[2], bI1[2], bH1[2], bI2[2], bH2[2], bE1[2], bE2[2];
#pragma unroll
  for (int tn = 0; tn < 2; ++tn) {
    int col = n16 + 16 * tn;
    bT1[tn] = (col < DDZC) ? bt1[col]   : 0.0f;
    bE1[tn] = (col < DDXC) ? be1[col]   : 0.0f;
    bE2[tn] = (col < DDXC) ? be2[col]   : 0.0f;
    bI1[tn] = (col < 30)   ? b_ih1[col] : 0.0f;
    bH1[tn] = (col < 30)   ? b_hh1[col] : 0.0f;
    bI2[tn] = (col < 30)   ? b_ih2[col] : 0.0f;
    bH2[tn] = (col < 30)   ? b_hh2[col] : 0.0f;
  }
  __syncthreads();

  // =================== sequential scan over T ===================
  for (int t = 0; t < T_LEN; ++t) {
    // --- stage x[t] (16 rows x 100 f32 -> f16 staging); prefetch next step ---
    {
      const float* row = x + ((long)(m0 + n16) * T_LEN + t) * DXC;
      int q0 = (lane < 16) ? 0 : 13;   // lane pairs split the 25 float4s of a row
#pragma unroll
      for (int i = 0; i < 13; ++i) {
        int q = q0 + i;
        if (q < 25) {
          float4 v = *(const float4*)(row + q * 4);
          union { _Float16 h[4]; uint2 u; } pk;
          pk.h[0] = (_Float16)v.x; pk.h[1] = (_Float16)v.y;
          pk.h[2] = (_Float16)v.z; pk.h[3] = (_Float16)v.w;
          *(uint2*)(xs + n16 * 136 + q * 4) = pk.u;
        }
      }
      if (t + 1 < T_LEN) {
        __builtin_prefetch(x   + ((long)(m0 + n16) * T_LEN + (t + 1)) * DXC, 0, 0);
        __builtin_prefetch(eps + ((long)(m0 + n16) * T_LEN + (t + 1)) * DZC, 0, 0);
      }
    }

    // --- stoch_trans hidden: hid = relu(h1 @ Wt1.T + bt1) ---
    v16h aH1 = loadA(h1s, 40, lane, 0);
    {
      v8f a0 = splat8(bT1[0]); a0 = wmma16(aH1, loadB(sB, F_WT1 + 0, lane), a0);
      v8f a1 = splat8(bT1[1]); a1 = wmma16(aH1, loadB(sB, F_WT1 + 1, lane), a1);
      storeD_h16(a0, hidS,      40, lane, true);
      storeD_h16(a1, hidS + 16, 40, lane, true);
    }

    // --- z = z_loc + softplus(z_scale_pre) * eps  (7 N-tiles of 16) ---
    {
      v16h aHid = loadA(hidS, 40, lane, 0);
#pragma unroll
      for (int tn = 0; tn < 7; ++tn) {
        v8f aL = splat8(bLoc[tn]); aL = wmma16(aHid, loadB(sB, F_WLOC + tn, lane), aL);
        v8f aS = splat8(bSc[tn]);  aS = wmma16(aHid, loadB(sB, F_WSC  + tn, lane), aS);
        int col = n16 + 16 * tn;
        float ev[8];
#pragma unroll
        for (int r = 0; r < 8; ++r) ev[r] = 0.0f;
        if (col < DZC) {
#pragma unroll
          for (int r = 0; r < 8; ++r)
            ev[r] = eps[((long)(m0 + mb + r) * T_LEN + t) * DZC + col];
        }
#pragma unroll
        for (int r = 0; r < 8; ++r) {
          float zv = aL[r] + softp(aS[r]) * ev[r];
          zs[(mb + r) * 136 + col] = (_Float16)zv;
        }
      }
    }

    // --- GRU1 gates: gi = z@W_ih1.T + b_ih1 (K=100, 4 slices); gh = h1@W_hh1.T + b_hh1 ---
    {
      v8f i0 = splat8(bI1[0]), i1 = splat8(bI1[1]);
#pragma unroll
      for (int s = 0; s < 4; ++s) {
        v16h az = loadA(zs, 136, lane, s);
        i0 = wmma16(az, loadB(sB, F_WIH1 + 2 * s + 0, lane), i0);
        i1 = wmma16(az, loadB(sB, F_WIH1 + 2 * s + 1, lane), i1);
      }
      v8f g0 = splat8(bH1[0]); g0 = wmma16(aH1, loadB(sB, F_WHH1 + 0, lane), g0);
      v8f g1 = splat8(bH1[1]); g1 = wmma16(aH1, loadB(sB, F_WHH1 + 1, lane), g1);
      storeD_f32(i0, gI,      33, lane); storeD_f32(i1, gI + 16, 33, lane);
      storeD_f32(g0, gH,      33, lane); storeD_f32(g1, gH + 16, 33, lane);
      gruCombine(gI, gH, h1f, h1s, dxs, lane);   // h1 <- h1n ; dxs[0:10) = h1n
    }

    // --- emitter: e = relu(relu(dx_in@We1.T+be1)@We2.T+be2); x_ps = sigmoid(e@We3.T+be3) ---
    {
      v16h aDx = loadA(dxs, 40, lane, 0);        // [h1n, h2_old] (shift semantics)
      v8f a0 = splat8(bE1[0]); a0 = wmma16(aDx, loadB(sB, F_WE1 + 0, lane), a0);
      v8f a1 = splat8(bE1[1]); a1 = wmma16(aDx, loadB(sB, F_WE1 + 1, lane), a1);
      storeD_h16(a0, es,      40, lane, true);
      storeD_h16(a1, es + 16, 40, lane, true);

      v16h aE = loadA(es, 40, lane, 0);
      v8f c0 = splat8(bE2[0]); c0 = wmma16(aE, loadB(sB, F_WE2 + 0, lane), c0);
      v8f c1 = splat8(bE2[1]); c1 = wmma16(aE, loadB(sB, F_WE2 + 1, lane), c1);
      storeD_h16(c0, es,      40, lane, true);
      storeD_h16(c1, es + 16, 40, lane, true);

      v16h aE2 = loadA(es, 40, lane, 0);
#pragma unroll
      for (int tn = 0; tn < 7; ++tn) {
        v8f acc = splat8(bE3[tn]);
        acc = wmma16(aE2, loadB(sB, F_WE3 + tn, lane), acc);
        int col = n16 + 16 * tn;
        if (col < DXC) {
#pragma unroll
          for (int r = 0; r < 8; ++r)
            out[((long)(m0 + mb + r) * T_LEN + t) * DXC + col] = sigm(acc[r]);
        }
      }
    }

    // --- GRU2 update with x[t] (after emitter consumed h2_old) ---
    {
      v8f i0 = splat8(bI2[0]), i1 = splat8(bI2[1]);
#pragma unroll
      for (int s = 0; s < 4; ++s) {
        v16h ax = loadA(xs, 136, lane, s);
        i0 = wmma16(ax, loadB(sB, F_WIH2 + 2 * s + 0, lane), i0);
        i1 = wmma16(ax, loadB(sB, F_WIH2 + 2 * s + 1, lane), i1);
      }
      v16h aH2 = loadA(h2s, 40, lane, 0);
      v8f g0 = splat8(bH2[0]); g0 = wmma16(aH2, loadB(sB, F_WHH2 + 0, lane), g0);
      v8f g1 = splat8(bH2[1]); g1 = wmma16(aH2, loadB(sB, F_WHH2 + 1, lane), g1);
      storeD_f32(i0, gI,      33, lane); storeD_f32(i1, gI + 16, 33, lane);
      storeD_f32(g0, gH,      33, lane); storeD_f32(g1, gH + 16, 33, lane);
      gruCombine(gI, gH, h2f, h2s, dxs + 10, lane);  // h2 <- h2n ; dxs[10:20) = h2n
    }
  }
}

extern "C" void kernel_launch(void* const* d_in, const int* in_sizes, int n_in,
                              void* d_out, int out_size, void* d_ws, size_t ws_size,
                              hipStream_t stream) {
  (void)in_sizes; (void)n_in; (void)out_size; (void)d_ws; (void)ws_size;
  const float* p[24];
  for (int i = 0; i < 24; ++i) p[i] = (const float*)d_in[i];
  dim3 grid(B_TOT / 16), block(32);   // one wave32 per 16-row batch tile, 16 WGPs
  ArFSSM_77103252898304_kernel<<<grid, block, 0, stream>>>(
      p[0], p[1], p[2], p[3], p[4], p[5], p[6], p[7], p[8], p[9], p[10], p[11],
      p[12], p[13], p[14], p[15], p[16], p[17], p[18], p[19], p[20], p[21], p[22], p[23],
      (float*)d_out);
}
